// SetConv_70325794505115
// MI455X (gfx1250) — matrix operand order repaired and checked by reading, not compiled, not run on hardware
//
#include <hip/hip_runtime.h>

// ---------- types for CDNA5 WMMA ----------
typedef __attribute__((ext_vector_type(16))) __bf16 bf16x16;
typedef __attribute__((ext_vector_type(8)))  float  f32x8;

union Frag {
  unsigned u[8];
  bf16x16  v;
};

__device__ __forceinline__ unsigned short f2bf(float f) {
  unsigned u = __float_as_uint(f);
  u += 0x7fffu + ((u >> 16) & 1u);           // round-to-nearest-even
  return (unsigned short)(u >> 16);
}
__device__ __forceinline__ unsigned pack2(float a, float b) {
  return (unsigned)f2bf(a) | ((unsigned)f2bf(b) << 16);
}
__device__ __forceinline__ f32x8 zero8() {
  f32x8 z;
#pragma unroll
  for (int r = 0; r < 8; ++r) z[r] = 0.0f;
  return z;
}

// Load A fragment (16x32 bf16 tile) from row-major bf16 LDS.
// ISA layout: lanes 0-15 (hi=0): V0-3 -> K=0..7, V4-7 -> K=16..23
//             lanes 16-31 (hi=1): V0-3 -> K=8..15, V4-7 -> K=24..31
__device__ __forceinline__ void load_afrag(const unsigned short* lds, int m, int stride,
                                           int kbase, int hi, Frag& fr) {
#pragma unroll
  for (int v = 0; v < 8; ++v) {
    int k = kbase + ((v < 4) ? (hi * 8 + 2 * v) : (16 + hi * 8 + 2 * (v - 4)));
    fr.u[v] = *(const unsigned*)(lds + m * stride + k);  // k even, stride even -> aligned
  }
}

// ---------- weight repack: W (c x n) -> bf16 pairs [k/2][n], W0 zero-padded 67->96 ----------
__global__ void prep_kernel(const float* __restrict__ W0, const float* __restrict__ W1,
                            const float* __restrict__ W2,
                            unsigned* __restrict__ W0p, unsigned* __restrict__ W1p,
                            unsigned* __restrict__ W2p) {
  int tid = blockIdx.x * blockDim.x + threadIdx.x;
  int nt  = gridDim.x * blockDim.x;
  for (int i = tid; i < 48 * 64; i += nt) {          // W0p: 96x64 -> 48x64 words
    int kk = i >> 6, n = i & 63;
    int k0 = 2 * kk, k1 = k0 + 1;
    float f0 = (k0 < 67) ? W0[k0 * 64 + n] : 0.0f;
    float f1 = (k1 < 67) ? W0[k1 * 64 + n] : 0.0f;
    W0p[i] = pack2(f0, f1);
  }
  for (int i = tid; i < 32 * 64; i += nt) {          // W1p: 64x64
    int kk = i >> 6, n = i & 63;
    W1p[i] = pack2(W1[(2 * kk) * 64 + n], W1[(2 * kk + 1) * 64 + n]);
  }
  for (int i = tid; i < 32 * 128; i += nt) {         // W2p: 64x128
    int kk = i >> 7, n = i & 127;
    W2p[i] = pack2(W2[(2 * kk) * 128 + n], W2[(2 * kk + 1) * 128 + n]);
  }
}

// ---------- farthest point sampling: one block per batch, point cloud in LDS ----------
__global__ __launch_bounds__(1024)
void fps_kernel(const float* __restrict__ xyz, float* __restrict__ new_xyz) {
  __shared__ float xyz_s[4096 * 3];
  __shared__ int   fps_s[1024];
  __shared__ float red_d[32];
  __shared__ int   red_i[32];
  __shared__ int   far_s;
  const int b = blockIdx.x;
  const int tid = threadIdx.x;
  const int lane = tid & 31, w = tid >> 5;
  const float* xb = xyz + (size_t)b * 4096 * 3;
  for (int e = tid; e < 4096 * 3; e += 1024) xyz_s[e] = xb[e];
  if (tid == 0) far_s = 0;
  __syncthreads();
  float px[4], py[4], pz[4], dd[4];
#pragma unroll
  for (int q = 0; q < 4; ++q) {
    int j = tid * 4 + q;
    px[q] = xyz_s[j * 3 + 0];
    py[q] = xyz_s[j * 3 + 1];
    pz[q] = xyz_s[j * 3 + 2];
    dd[q] = 1.0e10f;
  }
  for (int it = 0; it < 1024; ++it) {
    int far = far_s;
    if (tid == 0) fps_s[it] = far;       // record centroid used this iteration
    float cx = xyz_s[far * 3 + 0], cy = xyz_s[far * 3 + 1], cz = xyz_s[far * 3 + 2];
    float bd = -1.0f; int bi = 0;
#pragma unroll
    for (int q = 0; q < 4; ++q) {
      float dx = px[q] - cx, dy = py[q] - cy, dz = pz[q] - cz;
      float d = dx * dx + dy * dy + dz * dz;
      dd[q] = fminf(dd[q], d);
      if (dd[q] > bd) { bd = dd[q]; bi = tid * 4 + q; }
    }
#pragma unroll
    for (int off = 16; off > 0; off >>= 1) {     // wave32 argmax, min-index tiebreak
      float od = __shfl_xor(bd, off, 32);
      int   oi = __shfl_xor(bi, off, 32);
      if (od > bd || (od == bd && oi < bi)) { bd = od; bi = oi; }
    }
    if (lane == 0) { red_d[w] = bd; red_i[w] = bi; }
    __syncthreads();
    if (w == 0) {
      float bd2 = red_d[lane]; int bi2 = red_i[lane];
#pragma unroll
      for (int off = 16; off > 0; off >>= 1) {
        float od = __shfl_xor(bd2, off, 32);
        int   oi = __shfl_xor(bi2, off, 32);
        if (od > bd2 || (od == bd2 && oi < bi2)) { bd2 = od; bi2 = oi; }
      }
      if (lane == 0) far_s = bi2;
    }
    __syncthreads();
  }
  int j = fps_s[tid];
  float* o = new_xyz + ((size_t)b * 1024 + tid) * 3;
  o[0] = xyz_s[j * 3 + 0];
  o[1] = xyz_s[j * 3 + 1];
  o[2] = xyz_s[j * 3 + 2];
}

// ---------- ball query: one wave per centroid, first K=32 ascending indices ----------
__global__ __launch_bounds__(256)
void ball_kernel(const float* __restrict__ xyz, const float* __restrict__ new_xyz,
                 int* __restrict__ idx_out) {
  __shared__ int buf[8][32];
  const int lane = threadIdx.x & 31, w = threadIdx.x >> 5;
  const int g = blockIdx.x * 8 + w;
  const int b = g >> 10;
  const float* xb = xyz + (size_t)b * 4096 * 3;
  const float cx = new_xyz[g * 3 + 0], cy = new_xyz[g * 3 + 1], cz = new_xyz[g * 3 + 2];
  int cnt = 0;
  for (int base = 0; base < 4096; base += 32) {
    int j = base + lane;
    float dx = xb[j * 3 + 0] - cx, dy = xb[j * 3 + 1] - cy, dz = xb[j * 3 + 2] - cz;
    bool pred = (dx * dx + dy * dy + dz * dz) <= 0.04f;   // RADIUS^2
    unsigned mask = (unsigned)__ballot(pred);             // wave32 -> low 32 bits
    int pos = cnt + __popc(mask & ((1u << lane) - 1u));
    if (pred && pos < 32) buf[w][pos] = j;
    cnt += __popc(mask);
    if (cnt >= 32) break;
  }
  __syncthreads();
  int first = buf[w][0];                 // centroid itself guarantees cnt >= 1
  int val = (lane < cnt) ? buf[w][lane] : first;
  idx_out[(size_t)g * 32 + lane] = val;
}

// ---------- fused gather + 3-layer MLP (bf16 WMMA) + max-pool over K ----------
__global__ __launch_bounds__(256)
void mlp_kernel(const float* __restrict__ xyz, const float* __restrict__ pts,
                const unsigned* __restrict__ W0p, const unsigned* __restrict__ W1p,
                const unsigned* __restrict__ W2p,
                const float* __restrict__ b0, const float* __restrict__ b1,
                const float* __restrict__ b2,
                const float* __restrict__ new_xyz, const int* __restrict__ idx,
                float* __restrict__ out_pts) {
  __shared__ __align__(16) unsigned short a_lds[32 * 96];   // input tile, bf16, padded 67->96
  __shared__ __align__(16) unsigned short h_lds[32 * 64];   // hidden activations, bf16
  const int tid = threadIdx.x;
  const int lane = tid & 31, w = tid >> 5;
  const int hi = lane >> 4, nl = lane & 15;
  const int g = blockIdx.x, b = g >> 10;
  const int n64  = (w & 3) * 16 + nl;    // layer0/1 output column for this wave
  const int n128 = w * 16 + nl;          // layer2 output column for this wave

  // resident B fragments (weights), loaded once per wave
  Frag B0f[3], B1f[2], B2f[2];
#pragma unroll
  for (int f = 0; f < 3; ++f)
#pragma unroll
    for (int v = 0; v < 8; ++v)
      B0f[f].u[v] = W0p[(16 * f + 4 * hi + v + (v >= 4 ? 4 : 0)) * 64 + n64];
#pragma unroll
  for (int f = 0; f < 2; ++f)
#pragma unroll
    for (int v = 0; v < 8; ++v) {
      B1f[f].u[v] = W1p[(16 * f + 4 * hi + v + (v >= 4 ? 4 : 0)) * 64 + n64];
      B2f[f].u[v] = W2p[(16 * f + 4 * hi + v + (v >= 4 ? 4 : 0)) * 128 + n128];
    }
  const float bias0 = b0[n64], bias1 = b1[n64], bias2 = b2[n128];

  // gather group rows: [grouped_xyz - center (3) | grouped_points (64) | pad 0 (29)]
  const int*   ig = idx + (size_t)g * 32;
  const float* nx = new_xyz + (size_t)g * 3;
  for (int e = tid; e < 32 * 96; e += 256) {
    int k = e / 96, c = e - k * 96;
    int j = ig[k];
    float val = 0.0f;
    if (c < 3)       val = xyz[((size_t)b * 4096 + j) * 3 + c] - nx[c];
    else if (c < 67) val = pts[((size_t)b * 4096 + j) * 64 + (c - 3)];
    a_lds[e] = f2bf(val);
  }
  __syncthreads();

  const int mi = w >> 2;                 // M-tile (rows 0-15 or 16-31) for layers 0/1
  const int mrow = mi * 16 + nl;

  // layer 0: [32x96] @ [96x64]
  f32x8 acc = zero8();
#pragma unroll
  for (int f = 0; f < 3; ++f) {
    Frag a;
    load_afrag(a_lds, mrow, 96, 32 * f, hi, a);
    acc = __builtin_amdgcn_wmma_f32_16x16x32_bf16(false, a.v, false, B0f[f].v,
                                                  (short)0, acc, false, false);
  }
#pragma unroll
  for (int r = 0; r < 8; ++r) {
    float x = fmaxf(acc[r] + bias0, 0.0f);
    h_lds[(mi * 16 + hi * 8 + r) * 64 + n64] = f2bf(x);   // D layout -> row-major LDS
  }
  __syncthreads();

  // layer 1: [32x64] @ [64x64]
  Frag a1[2];
  load_afrag(h_lds, mrow, 64, 0,  hi, a1[0]);
  load_afrag(h_lds, mrow, 64, 32, hi, a1[1]);
  acc = zero8();
#pragma unroll
  for (int f = 0; f < 2; ++f)
    acc = __builtin_amdgcn_wmma_f32_16x16x32_bf16(false, a1[f].v, false, B1f[f].v,
                                                  (short)0, acc, false, false);
  __syncthreads();                       // all waves finished reading h0
#pragma unroll
  for (int r = 0; r < 8; ++r) {
    float x = fmaxf(acc[r] + bias1, 0.0f);
    h_lds[(mi * 16 + hi * 8 + r) * 64 + n64] = f2bf(x);
  }
  __syncthreads();

  // layer 2: [32x64] @ [64x128]; each wave does both M-tiles of its 16-col slice,
  // then max over all 32 rows (the K grouping axis)
  float pm = 0.0f;                       // post-ReLU values are >= 0
#pragma unroll
  for (int m2 = 0; m2 < 2; ++m2) {
    f32x8 acc2 = zero8();
    Frag a2;
    load_afrag(h_lds, m2 * 16 + nl, 64, 0, hi, a2);
    acc2 = __builtin_amdgcn_wmma_f32_16x16x32_bf16(false, a2.v, false, B2f[0].v,
                                                   (short)0, acc2, false, false);
    load_afrag(h_lds, m2 * 16 + nl, 64, 32, hi, a2);
    acc2 = __builtin_amdgcn_wmma_f32_16x16x32_bf16(false, a2.v, false, B2f[1].v,
                                                   (short)0, acc2, false, false);
#pragma unroll
    for (int r = 0; r < 8; ++r) pm = fmaxf(pm, fmaxf(acc2[r] + bias2, 0.0f));
  }
  pm = fmaxf(pm, __shfl_xor(pm, 16, 32));   // lane l and l^16 hold the same column
  if (lane < 16) out_pts[(size_t)g * 128 + w * 16 + nl] = pm;
}

extern "C" void kernel_launch(void* const* d_in, const int* in_sizes, int n_in,
                              void* d_out, int out_size, void* d_ws, size_t ws_size,
                              hipStream_t stream) {
  (void)in_sizes; (void)n_in; (void)out_size; (void)ws_size;
  const float* xyz = (const float*)d_in[0];
  const float* pts = (const float*)d_in[1];
  const float* W0  = (const float*)d_in[2];
  const float* b0  = (const float*)d_in[3];
  const float* W1  = (const float*)d_in[4];
  const float* b1  = (const float*)d_in[5];
  const float* W2  = (const float*)d_in[6];
  const float* b2  = (const float*)d_in[7];

  float* out      = (float*)d_out;
  float* new_xyz  = out;                                   // 16*1024*3 f32
  float* new_pts  = out + 16 * 1024 * 3;                   // 16*1024*128 f32
  int*   idx_out  = (int*)(out + 16 * 1024 * 3 + 16 * 1024 * 128); // 16*1024*32 i32

  unsigned* W0p = (unsigned*)d_ws;        // 48*64 words
  unsigned* W1p = W0p + 48 * 64;          // 32*64 words
  unsigned* W2p = W1p + 32 * 64;          // 32*128 words

  prep_kernel<<<8, 256, 0, stream>>>(W0, W1, W2, W0p, W1p, W2p);
  fps_kernel<<<16, 1024, 0, stream>>>(xyz, new_xyz);
  ball_kernel<<<2048, 256, 0, stream>>>(xyz, new_xyz, idx_out);
  mlp_kernel<<<16384, 256, 0, stream>>>(xyz, pts, W0p, W1p, W2p, b0, b1, b2,
                                        new_xyz, idx_out, new_pts);
}
